// KANLinear_66065186947649
// MI455X (gfx1250) — matrix operand order repaired and checked by reading, not compiled, not run on hardware
//
#include <hip/hip_runtime.h>
#include <hip/hip_bf16.h>
#include <stdint.h>

// ---------------------------------------------------------------------------
// KANLinear forward for MI455X (gfx1250, wave32, WMMA + Tensor Data Mover).
//   out = SiLU(x) @ Wb^T + einsum(bik,oik->bo)
// folded into one bf16 GEMM:  A(16384 x 4608) @ W(512 x 4608)^T, f32 accum.
// GEMM tiles are DMA'd into LDS by the TDM (tensor_load_to_lds), double
// buffered and overlapped with the WMMA pipeline; TENSORcnt + barriers sync.
// ---------------------------------------------------------------------------

#define BATCH       16384
#define IN_F        512
#define OUT_F       512
#define NSPLINE     8            // grid_size + spline_order
#define NB          9            // 1 (SiLU) + 8 spline bases
#define KDIM        (IN_F * NB)  // 4608
#define BM          128
#define BN          128
#define BK          64
#define NKT         (KDIM / BK)  // 72 K-tiles
#define LDS_STRIDE  (BK + 8)     // 72 bf16 = 144B rows (TDM pad reproduces this)

typedef __bf16 bf16x8  __attribute__((ext_vector_type(8)));
typedef __bf16 bf16x16 __attribute__((ext_vector_type(16)));
typedef float  f32x8   __attribute__((ext_vector_type(8)));
typedef unsigned int tdm_u32x4 __attribute__((ext_vector_type(4)));
typedef int          tdm_i32x8 __attribute__((ext_vector_type(8)));
typedef int          tdm_i32x4 __attribute__((ext_vector_type(4)));

__device__ __forceinline__ __bf16 f2bf(float f) {
    union { float f; uint32_t u; } v; v.f = f;
    uint32_t r = v.u + 0x7FFFu + ((v.u >> 16) & 1u);   // round-to-nearest-even
    uint16_t h = (uint16_t)(r >> 16);
    return __builtin_bit_cast(__bf16, h);
}

// ---------------------------------------------------------------------------
// TDM: DMA one 128-row x 64-col bf16 tile (row pitch KDIM) into LDS,
// inserting 16B of padding after every 128B row => LDS stride 72 bf16.
// D# bitfields per CDNA5 ISA sec 8.3/8.4. 2-D tile: groups 2/3 zero.
// This toolchain exposes the 6-arg builtin form (extra zero-filled i32x8).
// ---------------------------------------------------------------------------
__device__ __forceinline__ void tdm_load_tile(uint32_t lds_off, const void* gsrc) {
    const uint64_t ga = (uint64_t)(uintptr_t)gsrc;
    tdm_u32x4 g0;
    g0[0] = 1u;                                        // count=1, user desc
    g0[1] = lds_off;                                   // lds_addr
    g0[2] = (uint32_t)ga;                              // global_addr[31:0]
    g0[3] = (uint32_t)((ga >> 32) & 0x01FFFFFFu)       // global_addr[56:32]
          | (2u << 30);                                // type=2 ("image")
    tdm_i32x8 g1;
    g1[0] = (int)((1u << 16)                           // data_size = 2 bytes
                | (1u << 20)                           // pad_enable
                | (4u << 22)                           // pad_interval: 32 DW (128B)
                | (3u << 25));                         // pad_amount: 4 DW (16B)
    g1[1] = (int)((uint32_t)BK << 16);                 // tensor_dim0 = 64
    g1[2] = (int)((uint32_t)BM << 16);                 // tensor_dim1 = 128
    g1[3] = (int)((uint32_t)BK << 16);                 // tile_dim0 = 64
    g1[4] = (int)BM;                                   // tile_dim1 = 128 (dim2=0)
    g1[5] = (int)KDIM;                                 // tensor_dim0_stride lo
    g1[6] = 0;                                         // stride hi / dim1_stride lo
    g1[7] = 0;
    tdm_i32x4 gz4; gz4[0] = 0; gz4[1] = 0; gz4[2] = 0; gz4[3] = 0;
    tdm_i32x8 gz8;
#pragma unroll
    for (int q = 0; q < 8; ++q) gz8[q] = 0;
    __builtin_amdgcn_tensor_load_to_lds(g0, g1, gz4, gz4, gz8, 0);
}

// ---------------------------------------------------------------------------
// Kernel 1: per (b,i) compute SiLU(x) and the 8 cubic B-spline bases
// (Cox-de Boor, fixed unroll), write bf16 activations A[b, j*IN_F + i].
// ---------------------------------------------------------------------------
__global__ __launch_bounds__(256) void kan_pack_act(
    const float* __restrict__ x, const float* __restrict__ grid,
    __bf16* __restrict__ A)
{
    const int idx = blockIdx.x * 256 + threadIdx.x;          // over BATCH*IN_F
    const int b = idx / IN_F;
    const int i = idx - b * IN_F;
    const float v = x[(size_t)b * IN_F + i];

    float g[12];
#pragma unroll
    for (int t = 0; t < 12; ++t) g[t] = grid[t];             // row 0, uniform

    float bs[11];
#pragma unroll
    for (int t = 0; t < 11; ++t)
        bs[t] = (v >= g[t] && v < g[t + 1]) ? 1.0f : 0.0f;

#pragma unroll
    for (int k = 1; k <= 3; ++k) {
#pragma unroll
        for (int t = 0; t < 11 - k; ++t) {
            float left  = (v - g[t])         / (g[t + k]     - g[t])     * bs[t];
            float right = (g[t + k + 1] - v) / (g[t + k + 1] - g[t + 1]) * bs[t + 1];
            bs[t] = left + right;
        }
    }

    const float silu = v / (1.0f + __expf(-v));

    const size_t base = (size_t)b * KDIM + i;   // j-major channel layout
    A[base] = f2bf(silu);
#pragma unroll
    for (int t = 0; t < NSPLINE; ++t)
        A[base + (size_t)(t + 1) * IN_F] = f2bf(bs[t]);
}

// ---------------------------------------------------------------------------
// Kernel 2: pack weights to bf16, layout matching A: W[o, j*IN_F + i]
// ---------------------------------------------------------------------------
__global__ __launch_bounds__(256) void kan_pack_w(
    const float* __restrict__ bw, const float* __restrict__ sw,
    const float* __restrict__ ss, __bf16* __restrict__ W)
{
    const int idx = blockIdx.x * 256 + threadIdx.x;          // over OUT_F*KDIM
    const int o = idx / KDIM;
    const int k = idx - o * KDIM;
    const int j = k / IN_F;
    const int i = k - j * IN_F;
    const size_t oi = (size_t)o * IN_F + i;
    float w = (j == 0) ? bw[oi] : sw[oi * NSPLINE + (j - 1)] * ss[oi];
    W[(size_t)o * KDIM + k] = f2bf(w);
}

// ---------------------------------------------------------------------------
// Kernel 3: C = A x W^T. 8 waves in 2(M)x4(N); wave tile 64x32.
// TDM double-buffered LDS staging; 16 v_wmma_f32_16x16x32_bf16 per K-tile
// with all 12 fragments hoisted before the WMMA burst.
// ---------------------------------------------------------------------------
__global__ __launch_bounds__(256) void kan_gemm(
    const __bf16* __restrict__ A, const __bf16* __restrict__ W,
    float* __restrict__ C)
{
    __shared__ __bf16 lA[2][BM][LDS_STRIDE];
    __shared__ __bf16 lB[2][BN][LDS_STRIDE];

    const int tid  = threadIdx.x;
    const int lane = tid & 31;
    const int wave = tid >> 5;
    const int wm   = wave >> 2;              // 0..1 -> 64-row slab
    const int wn   = wave & 3;               // 0..3 -> 32-col slab
    const int block_m = blockIdx.y * BM;
    const int block_n = blockIdx.x * BN;

    f32x8 acc[4][2] = {};                    // 4 M-subtiles x 2 N-subtiles

    const int hsel = lane >> 4;              // 0: lanes 0-15, 1: lanes 16-31
    const int fr   = lane & 15;              // row/col within 16x16 tile

    // TDM issuer: wave 0 only, via a *scalar* branch (TDM ignores EXEC).
    const bool issuer = (__builtin_amdgcn_readfirstlane(tid) < 32);
    const uint32_t ldsA0 = (uint32_t)(uintptr_t)&lA[0][0][0];
    const uint32_t ldsB0 = (uint32_t)(uintptr_t)&lB[0][0][0];
    const uint32_t bufBytes = BM * LDS_STRIDE * (uint32_t)sizeof(__bf16);
    const __bf16* gA = A + (size_t)block_m * KDIM;
    const __bf16* gB = W + (size_t)block_n * KDIM;

    if (issuer) {                            // prefetch tile 0 into buffer 0
        tdm_load_tile(ldsA0, gA);
        tdm_load_tile(ldsB0, gB);
    }

    for (int it = 0; it < NKT; ++it) {
        const int cur = it & 1;
        if (issuer) {
            if (it + 1 < NKT) {              // prefetch next tile into other buf
                const int nxt = cur ^ 1;
                tdm_load_tile(ldsA0 + nxt * bufBytes, gA + (size_t)(it + 1) * BK);
                tdm_load_tile(ldsB0 + nxt * bufBytes, gB + (size_t)(it + 1) * BK);
                __builtin_amdgcn_s_wait_tensorcnt(2);  // current pair landed
            } else {
                __builtin_amdgcn_s_wait_tensorcnt(0);
            }
        }
        __syncthreads();                     // LDS tile `cur` visible to all

        // ---- hoist all 12 fragments of this BK=64 tile, then 16 WMMAs ----
        bf16x16 af[2][4];                    // [ks][m-subtile]
        bf16x16 bfr[2][2];                   // [ks][n-subtile]
#pragma unroll
        for (int ks2 = 0; ks2 < 2; ++ks2) {
            const int ks = ks2 * 32;
#pragma unroll
            for (int mt = 0; mt < 4; ++mt) { // A: lanes 0-15 K{0-7,16-23}, 16-31 K{8-15,24-31}
                const int row = wm * 64 + mt * 16 + fr;
                const int c0  = ks + hsel * 8;
                union { bf16x16 v; bf16x8 h[2]; } u;
                u.h[0] = *(const bf16x8*)&lA[cur][row][c0];
                u.h[1] = *(const bf16x8*)&lA[cur][row][c0 + 16];
                af[ks2][mt] = u.v;
            }
#pragma unroll
            for (int nt = 0; nt < 2; ++nt) { // B: lanes 0-15 K{0-7,16-23}, 16-31 K{8-15,24-31}
                const int col = wn * 32 + nt * 16 + fr;
                const int c0  = ks + hsel * 8;
                union { bf16x16 v; bf16x8 h[2]; } u;
                u.h[0] = *(const bf16x8*)&lB[cur][col][c0];
                u.h[1] = *(const bf16x8*)&lB[cur][col][c0 + 16];
                bfr[ks2][nt] = u.v;
            }
        }
#pragma unroll
        for (int ks2 = 0; ks2 < 2; ++ks2)
#pragma unroll
            for (int mt = 0; mt < 4; ++mt)
#pragma unroll
                for (int nt = 0; nt < 2; ++nt)
                    acc[mt][nt] = __builtin_amdgcn_wmma_f32_16x16x32_bf16(
                        false, af[ks2][mt], false, bfr[ks2][nt],
                        (short)0, acc[mt][nt], false, false);

        __syncthreads();                     // done reading `cur` before refill
    }

    // Epilogue: C layout per ISA: VGPR r -> M = r + 8*hsel, N = lane&15
#pragma unroll
    for (int mt = 0; mt < 4; ++mt)
#pragma unroll
        for (int nt = 0; nt < 2; ++nt) {
            const int col   = block_n + wn * 32 + nt * 16 + fr;
            const int rbase = block_m + wm * 64 + mt * 16 + hsel * 8;
#pragma unroll
            for (int r = 0; r < 8; ++r)
                C[(size_t)(rbase + r) * OUT_F + col] = acc[mt][nt][r];
        }
}

// ---------------------------------------------------------------------------
// Host launcher. d_ws holds A_bf16 (16384*4608*2 = 151 MB, L2-resident) then
// W_bf16 (4.7 MB).
// ---------------------------------------------------------------------------
extern "C" void kernel_launch(void* const* d_in, const int* in_sizes, int n_in,
                              void* d_out, int out_size, void* d_ws, size_t ws_size,
                              hipStream_t stream) {
    const float* x    = (const float*)d_in[0];
    const float* grid = (const float*)d_in[1];
    const float* bw   = (const float*)d_in[2];
    const float* sw   = (const float*)d_in[3];
    const float* ss   = (const float*)d_in[4];
    float* out        = (float*)d_out;

    __bf16* Abf = (__bf16*)d_ws;
    __bf16* Wbf = Abf + (size_t)BATCH * KDIM;

    kan_pack_act<<<(BATCH * IN_F) / 256, 256, 0, stream>>>(x, grid, Abf);
    kan_pack_w<<<(OUT_F * KDIM) / 256, 256, 0, stream>>>(bw, sw, ss, Wbf);
    kan_gemm<<<dim3(OUT_F / BN, BATCH / BM), 256, 0, stream>>>(Abf, Wbf, out);
}
